// CustomNodeGCN_3908420239972
// MI455X (gfx1250) — matrix operand order repaired and checked by reading, not compiled, not run on hardware
//
#include <hip/hip_runtime.h>
#include <hip/hip_bf16.h>

#define NN   50000
#define EE   800000
#define NPAD 50048            // NN rounded up to multiple of 128 (GEMM tile rows)
#define EPSF 1e-5f

typedef __attribute__((ext_vector_type(16))) _Float16 v16h;
typedef __attribute__((ext_vector_type(8)))  _Float16 v8h;
typedef __attribute__((ext_vector_type(8)))  float    v8f;

// ---------------------------------------------------------------------------
// Degree / normalization
// ---------------------------------------------------------------------------
__global__ void init_deg_kernel(float* __restrict__ deg, int n) {
    int i = blockIdx.x * 256 + threadIdx.x;
    if (i < n) deg[i] = 1.0f;                      // self loop contribution
}

__global__ void deg_accum_kernel(const int* __restrict__ dst, float* __restrict__ deg, int e) {
    int i = blockIdx.x * 256 + threadIdx.x;
    if (i < e) atomicAdd(&deg[dst[i]], 1.0f);
}

__global__ void dinv_kernel(float* __restrict__ deg, int n) {
    int i = blockIdx.x * 256 + threadIdx.x;
    if (i < n) deg[i] = rsqrtf(deg[i]);            // deg >= 1 always
}

// ---------------------------------------------------------------------------
// f32 -> f16 conversion (producer for WMMA A-operands)
// ---------------------------------------------------------------------------
__global__ void convert_f16_kernel(const float* __restrict__ in,
                                   _Float16* __restrict__ out16, size_t n) {
    size_t i = (size_t)blockIdx.x * 256 + threadIdx.x;
    if (i < n) out16[i] = (_Float16)in[i];
}

// ---------------------------------------------------------------------------
// WMMA GEMM: out[n x FOUT] = act(A16[n x 128] @ W[128 x FOUT] (+ bias))
//   A16 : f16 activations, row-major, padded to >= gridDim.x*128 rows
//   A-tile (32KB, contiguous) streamed to LDS via global_load_async_to_lds_b128
//   W    : f32, transposed+converted into LDS so each B fragment is two runs of 8
//          contiguous f16 (two ds_load_b128 per fragment)
//   OUT16=true  -> writes f16 (feeds another GEMM)
//   OUT16=false -> writes f32
// ---------------------------------------------------------------------------
template<int FOUT, bool RELU, bool BIAS, bool OUT16>
__global__ __launch_bounds__(256)
void gemm_f16_wmma(const _Float16* __restrict__ A16, const float* __restrict__ W,
                   const float* __restrict__ bias, void* __restrict__ outv, int nrows)
{
    __shared__ _Float16 At[128 * 128];     // At[m][k]  activation tile (32KB)
    __shared__ _Float16 Wt[FOUT * 128];    // Wt[n][k]  transposed weights

    const int tid     = threadIdx.x;
    const int rowBase = blockIdx.x * 128;

    // --- async DMA: 32KB contiguous A-tile -> LDS (ASYNCcnt path) ---
    {
        const unsigned long long gbase =
            (unsigned long long)(uintptr_t)(A16 + (size_t)rowBase * 128);
        const unsigned ldsA = (unsigned)(uintptr_t)At;   // LDS offset in low 32 bits
#pragma unroll
        for (int i = 0; i < 8; ++i) {
            unsigned off = (unsigned)tid * 16u + (unsigned)i * 4096u;
            unsigned la  = ldsA + off;
            unsigned long long ga = gbase + off;
            asm volatile("global_load_async_to_lds_b128 %0, %1, off"
                         :: "v"(la), "v"(ga) : "memory");
        }
    }

    // --- stage weights transposed as f16 (coalesced f32 reads) ---
    for (int i = tid; i < FOUT * 128; i += 256) {
        int k = i / FOUT, n = i % FOUT;
        Wt[n * 128 + k] = (_Float16)W[k * FOUT + n];
    }

    asm volatile("s_wait_asynccnt 0x0" ::: "memory");
    __syncthreads();

    const int wave  = tid >> 5;
    const int lane  = tid & 31;
    const int sel   = lane >> 4;     // which K-half this lane holds
    const int l16   = lane & 15;
    const int mBase = wave * 16;
    const bool full = (rowBase + 128) <= nrows;

    // Preload the wave's A fragments (held in VGPRs across all N tiles).
    v16h afrag[4];
#pragma unroll
    for (int kt = 0; kt < 4; ++kt) {
        const int kBase = kt * 32;
        const int arow  = mBase + l16;
        v8h alo = *(const v8h*)&At[arow * 128 + kBase + sel * 8];
        v8h ahi = *(const v8h*)&At[arow * 128 + kBase + 16 + sel * 8];
#pragma unroll
        for (int i = 0; i < 8; ++i) { afrag[kt][i] = alo[i]; afrag[kt][i + 8] = ahi[i]; }
    }

    for (int nt = 0; nt < FOUT / 16; ++nt) {
        const int nBase = nt * 16;
        v8f acc = {};
#pragma unroll
        for (int kt = 0; kt < 4; ++kt) {
            const int kBase = kt * 32;
            v8h blo = *(const v8h*)&Wt[(nBase + l16) * 128 + kBase + sel * 8];
            v8h bhi = *(const v8h*)&Wt[(nBase + l16) * 128 + kBase + 16 + sel * 8];
            v16h b;
#pragma unroll
            for (int i = 0; i < 8; ++i) { b[i] = blo[i]; b[i + 8] = bhi[i]; }
            acc = __builtin_amdgcn_wmma_f32_16x16x32_f16(
                false, afrag[kt], false, b, (short)0, acc, false, false);
        }
        const int   col = nBase + l16;
        const float bv  = BIAS ? bias[col] : 0.f;

        // C layout: VGPR r, lanes 0-15 -> M=r, lanes 16-31 -> M=8+r, N=l16
        if (OUT16) {
            _Float16* out = (_Float16*)outv;
            if (full) {
#pragma unroll
                for (int r = 0; r < 8; ++r) {
                    int grow = rowBase + mBase + r + sel * 8;
                    float v = acc[r] + bv;
                    if (RELU) v = v > 0.f ? v : 0.f;
                    out[(size_t)grow * FOUT + col] = (_Float16)v;
                }
            } else {
#pragma unroll
                for (int r = 0; r < 8; ++r) {
                    int grow = rowBase + mBase + r + sel * 8;
                    float v = acc[r] + bv;
                    if (RELU) v = v > 0.f ? v : 0.f;
                    if (grow < nrows) out[(size_t)grow * FOUT + col] = (_Float16)v;
                }
            }
        } else {
            float* out = (float*)outv;
            if (full) {
#pragma unroll
                for (int r = 0; r < 8; ++r) {
                    int grow = rowBase + mBase + r + sel * 8;
                    float v = acc[r] + bv;
                    if (RELU) v = v > 0.f ? v : 0.f;
                    out[(size_t)grow * FOUT + col] = v;
                }
            } else {
#pragma unroll
                for (int r = 0; r < 8; ++r) {
                    int grow = rowBase + mBase + r + sel * 8;
                    float v = acc[r] + bv;
                    if (RELU) v = v > 0.f ? v : 0.f;
                    if (grow < nrows) out[(size_t)grow * FOUT + col] = v;
                }
            }
        }
    }
}

// ---------------------------------------------------------------------------
// agg[i][f] = bias[f] + dinv[i]^2 * hw[i][f]   (self-loop edge folded in)
// ---------------------------------------------------------------------------
__global__ void agg_init_kernel(float* __restrict__ agg, const float* __restrict__ hw,
                                const float* __restrict__ dinv,
                                const float* __restrict__ bias, int n)
{
    size_t i = (size_t)blockIdx.x * 256 + threadIdx.x;
    if (i >= (size_t)n * 128) return;
    int row = (int)(i >> 7), f = (int)(i & 127);
    float di = dinv[row];
    agg[i] = bias[f] + di * di * hw[i];
}

// ---------------------------------------------------------------------------
// Scatter: one wave per edge, each lane handles 4 features (float4).
// ---------------------------------------------------------------------------
__global__ __launch_bounds__(256)
void scatter_kernel(const int* __restrict__ src, const int* __restrict__ dst,
                    const float* __restrict__ dinv, const float* __restrict__ h,
                    float* __restrict__ agg, int nedges)
{
    size_t t = (size_t)blockIdx.x * 256 + threadIdx.x;
    int e    = (int)(t >> 5);
    int lane = (int)(t & 31);
    if (e >= nedges) return;
    int s = src[e], d = dst[e];
    if (e + 512 < nedges) {                       // warm L2 for a future gather row
        int s2 = src[e + 512];
        __builtin_prefetch(&h[(size_t)s2 * 128 + lane * 4], 0, 1);
    }
    float nrm = dinv[s] * dinv[d];
    const float4 v = ((const float4*)(h + (size_t)s * 128))[lane];
    float* o = agg + (size_t)d * 128 + (size_t)lane * 4;
    atomicAdd(o + 0, v.x * nrm);
    atomicAdd(o + 1, v.y * nrm);
    atomicAdd(o + 2, v.z * nrm);
    atomicAdd(o + 3, v.w * nrm);
}

// ---------------------------------------------------------------------------
// BatchNorm (biased var) + ReLU; apply writes the f16 activation for next GEMM
// ---------------------------------------------------------------------------
__global__ void zero_stats_kernel(float* __restrict__ p) { p[threadIdx.x] = 0.f; }

__global__ __launch_bounds__(128)
void bn_stats_kernel(const float* __restrict__ x, float* __restrict__ stats, int n)
{
    int f = threadIdx.x;                          // 128 features
    float s = 0.f, s2 = 0.f;
    for (int r = blockIdx.x; r < n; r += gridDim.x) {
        float v = x[(size_t)r * 128 + f];
        s += v; s2 += v * v;
    }
    atomicAdd(&stats[f], s);
    atomicAdd(&stats[128 + f], s2);
}

__global__ void bn_apply_f16_kernel(const float* __restrict__ x,
                                    const float* __restrict__ stats,
                                    const float* __restrict__ g,
                                    const float* __restrict__ b,
                                    _Float16* __restrict__ out16, int n)
{
    size_t i = (size_t)blockIdx.x * 256 + threadIdx.x;
    if (i >= (size_t)n * 128) return;
    int f = (int)(i & 127);
    float inv_n = 1.f / (float)n;
    float m   = stats[f] * inv_n;
    float var = stats[128 + f] * inv_n - m * m;
    float y = (x[i] - m) * rsqrtf(var + EPSF) * g[f] + b[f];
    out16[i] = (_Float16)(y > 0.f ? y : 0.f);
}

// ---------------------------------------------------------------------------
extern "C" void kernel_launch(void* const* d_in, const int* in_sizes, int n_in,
                              void* d_out, int out_size, void* d_ws, size_t ws_size,
                              hipStream_t stream)
{
    (void)in_sizes; (void)n_in; (void)out_size; (void)ws_size;

    const float* x       = (const float*)d_in[0];
    const int*   ei      = (const int*)  d_in[1];
    const int*   src     = ei;
    const int*   dst     = ei + EE;
    const float* pre_w1  = (const float*)d_in[2];
    const float* pre_b1  = (const float*)d_in[3];
    const float* pre_w2  = (const float*)d_in[4];
    const float* pre_b2  = (const float*)d_in[5];
    const float* conv_w0 = (const float*)d_in[6];
    const float* conv_b0 = (const float*)d_in[7];
    const float* conv_w1 = (const float*)d_in[8];
    const float* conv_b1 = (const float*)d_in[9];
    const float* conv_w2 = (const float*)d_in[10];
    const float* conv_b2 = (const float*)d_in[11];
    const float* bn_g0   = (const float*)d_in[12];
    const float* bn_b0   = (const float*)d_in[13];
    const float* bn_g1   = (const float*)d_in[14];
    const float* bn_b1   = (const float*)d_in[15];
    const float* post_w1 = (const float*)d_in[16];
    const float* post_b1 = (const float*)d_in[17];
    const float* post_w2 = (const float*)d_in[18];
    const float* post_b2 = (const float*)d_in[19];
    float* out = (float*)d_out;

    // ---- workspace layout ----
    char* w = (char*)d_ws;
    float* bufA = (float*)w; w += sizeof(float) * (size_t)NPAD * 128;   // HW (f32)
    float* bufB = (float*)w; w += sizeof(float) * (size_t)NPAD * 128;   // agg (f32)
    float* bufC = (float*)w; w += sizeof(float) * (size_t)NPAD * 128;   // agg (f32)
    float* dinv = (float*)w; w += sizeof(float) * (size_t)NPAD;
    float* stats = (float*)w; w += sizeof(float) * 256;
    _Float16* X16  = (_Float16*)w; w += sizeof(_Float16) * (size_t)NPAD * 128;
    _Float16* H16a = (_Float16*)w; w += sizeof(_Float16) * (size_t)NPAD * 128;
    _Float16* H16b = (_Float16*)w; w += sizeof(_Float16) * (size_t)NPAD * 128;

    const int gN    = (NN + 255) / 256;
    const int gE    = (EE + 255) / 256;
    const int gElem = (int)(((size_t)NN * 128 + 255) / 256);
    const int gScat = (int)(((size_t)EE * 32 + 255) / 256);
    const int gGemm = NPAD / 128;   // 391

    // --- normalization ---
    init_deg_kernel<<<gN, 256, 0, stream>>>(dinv, NN);
    deg_accum_kernel<<<gE, 256, 0, stream>>>(dst, dinv, EE);
    dinv_kernel<<<gN, 256, 0, stream>>>(dinv, NN);

    // --- pre-MLP (f16 chain) ---
    convert_f16_kernel<<<gElem, 256, 0, stream>>>(x, X16, (size_t)NN * 128);
    gemm_f16_wmma<128, true, true, true ><<<gGemm, 256, 0, stream>>>(X16,  pre_w1, pre_b1, H16a, NN);
    gemm_f16_wmma<128, true, true, true ><<<gGemm, 256, 0, stream>>>(H16a, pre_w2, pre_b2, H16b, NN);

    // --- conv0: H16b -> bufC -> H16a ---
    gemm_f16_wmma<128, false, false, false><<<gGemm, 256, 0, stream>>>(H16b, conv_w0, nullptr, bufA, NN);
    agg_init_kernel<<<gElem, 256, 0, stream>>>(bufC, bufA, dinv, conv_b0, NN);
    scatter_kernel<<<gScat, 256, 0, stream>>>(src, dst, dinv, bufA, bufC, EE);
    zero_stats_kernel<<<1, 256, 0, stream>>>(stats);
    bn_stats_kernel<<<256, 128, 0, stream>>>(bufC, stats, NN);
    bn_apply_f16_kernel<<<gElem, 256, 0, stream>>>(bufC, stats, bn_g0, bn_b0, H16a, NN);

    // --- conv1: H16a -> bufB -> H16b ---
    gemm_f16_wmma<128, false, false, false><<<gGemm, 256, 0, stream>>>(H16a, conv_w1, nullptr, bufA, NN);
    agg_init_kernel<<<gElem, 256, 0, stream>>>(bufB, bufA, dinv, conv_b1, NN);
    scatter_kernel<<<gScat, 256, 0, stream>>>(src, dst, dinv, bufA, bufB, EE);
    zero_stats_kernel<<<1, 256, 0, stream>>>(stats);
    bn_stats_kernel<<<256, 128, 0, stream>>>(bufB, stats, NN);
    bn_apply_f16_kernel<<<gElem, 256, 0, stream>>>(bufB, stats, bn_g1, bn_b1, H16b, NN);

    // --- conv2: H16b -> bufC (no BN) ---
    gemm_f16_wmma<128, false, false, false><<<gGemm, 256, 0, stream>>>(H16b, conv_w2, nullptr, bufA, NN);
    agg_init_kernel<<<gElem, 256, 0, stream>>>(bufC, bufA, dinv, conv_b2, NN);
    scatter_kernel<<<gScat, 256, 0, stream>>>(src, dst, dinv, bufA, bufC, EE);

    // --- post-MLP ---
    convert_f16_kernel<<<gElem, 256, 0, stream>>>(bufC, H16a, (size_t)NN * 128);
    gemm_f16_wmma<128, true, true, true ><<<gGemm, 256, 0, stream>>>(H16a, post_w1, post_b1, H16b, NN);
    gemm_f16_wmma<64, false, true, false><<<gGemm, 256, 0, stream>>>(H16b, post_w2, post_b2, out, NN);
}
